// PMWA_24842090840472
// MI455X (gfx1250) — matrix operand order repaired and checked by reading, not compiled, not run on hardware
//
#include <hip/hip_runtime.h>
#include <hip/hip_bf16.h>

#define D 128
#define TILE 16
#define SIGMA 0.1f
#define EPS 1e-12f

typedef _Float16 v8h  __attribute__((ext_vector_type(8)));
typedef _Float16 v16h __attribute__((ext_vector_type(16)));
typedef float    v8f  __attribute__((ext_vector_type(8)));
typedef int      v4i  __attribute__((vector_size(16)));

union V16 { v16h v; v8h h[2]; };

#if defined(__AMDGCN__) && __has_builtin(__builtin_amdgcn_global_load_async_to_lds_b128)
#define USE_ASYNC_LDS 1
// async DMA global(AS1) -> LDS(AS3); staged casts: addrspacecast on void, then retype
__device__ __forceinline__ void async_copy_b128(const float* g, float* l) {
    const void* gv = (const void*)g;
    void*       lv = (void*)l;
    __attribute__((address_space(1))) void* g1 =
        (__attribute__((address_space(1))) void*)(const __attribute__((address_space(1))) void*)gv;
    __attribute__((address_space(3))) void* l3 = (__attribute__((address_space(3))) void*)lv;
    __builtin_amdgcn_global_load_async_to_lds_b128(
        (__attribute__((address_space(1))) v4i*)g1,
        (__attribute__((address_space(3))) v4i*)l3, 0, 0);
}
#else
#define USE_ASYNC_LDS 0
#endif

__device__ __forceinline__ void wait_asynccnt0() {
#if __has_builtin(__builtin_amdgcn_s_wait_asynccnt)
    __builtin_amdgcn_s_wait_asynccnt(0);
#elif defined(__AMDGCN__)
    asm volatile("s_wait_asynccnt 0x0" ::: "memory");
#endif
}

__device__ __forceinline__ float fast_rcp(float x) {
#if defined(__AMDGCN__)
    return __builtin_amdgcn_rcpf(x);   // v_rcp_f32, 1 ulp
#else
    return 1.0f / x;
#endif
}

// ---------------- RNG: deterministic counter-based gaussian ----------------
__device__ __forceinline__ unsigned pcg_hash(unsigned v) {
    unsigned state = v * 747796405u + 2891336453u;
    unsigned word  = ((state >> ((state >> 28u) + 4u)) ^ state) * 277803737u;
    return (word >> 22u) ^ word;
}

__device__ __forceinline__ float gaussf(unsigned idx, unsigned seed) {
    unsigned a = pcg_hash(idx ^ (seed * 0x9E3779B9u));
    unsigned b = pcg_hash(a + 0x85EBCA6Bu);
    float u1 = fmaxf((float)a * 2.3283064365386963e-10f, 1e-12f); // (0,1]
    float u2 = (float)b * 2.3283064365386963e-10f;
    return sqrtf(-2.0f * __logf(u1)) * __cosf(6.28318530717958647f * u2);
}

__device__ __forceinline__ void atomic_add_f32(float* p, float v) {
    // relaxed + agent scope -> hardware global_atomic_add_f32 (no CAS loop)
    __hip_atomic_fetch_add(p, v, __ATOMIC_RELAXED, __HIP_MEMORY_SCOPE_AGENT);
}

// ---------------- zero a buffer (float4 granularity) ----------------
__global__ __launch_bounds__(256) void zero_kernel(float4* __restrict__ p, long long n4) {
    long long i = (long long)blockIdx.x * blockDim.x + threadIdx.x;
    if (i < n4) p[i] = make_float4(0.f, 0.f, 0.f, 0.f);
}

// ---------------- L2-normalize rows: one wave (32 lanes) per 128-float row ----------------
__global__ __launch_bounds__(256) void norm_kernel(const float* __restrict__ in,
                                                   float* __restrict__ out, int N) {
    int row  = blockIdx.x * 8 + (threadIdx.x >> 5);
    int lane = threadIdx.x & 31;
    if (row >= N) return;
    long long base = (long long)row * D + lane * 4;
    float4 v = *(const float4*)(in + base);
    float s = v.x * v.x + v.y * v.y + v.z * v.z + v.w * v.w;
#pragma unroll
    for (int off = 16; off > 0; off >>= 1) s += __shfl_xor(s, off);
    float inv = fast_rcp(fmaxf(sqrtf(s), EPS));
    v.x *= inv; v.y *= inv; v.z *= inv; v.w *= inv;
    *(float4*)(out + base) = v;
}

// ---------------- per-hop edge kernel: 1 wave handles 16 edges via WMMA ----------------
// dots(16 edges) = diag( A(16x128 src rows) x B^T(16x128 dst rows) ), f16 in / f32 acc.
__global__ __launch_bounds__(32) void edge_kernel(const float* __restrict__ h,
                                                  const int* __restrict__ src,
                                                  const int* __restrict__ dst,
                                                  float* __restrict__ aggr, int E) {
    __shared__ float    sA [TILE][D + 4];   // src rows f32 (scatter + f16 convert), padded
    __shared__ float    sB [TILE][D + 4];   // dst rows f32 (f16 convert), padded
    __shared__ _Float16 sAh[TILE][D + 8];   // src rows f16 (WMMA A), 16B-aligned row stride
    __shared__ _Float16 sBh[TILE][D + 8];   // dst rows f16 (WMMA B)
    __shared__ float    sC [8][32];         // accumulator spill for diagonal extraction
    __shared__ float    sAlpha[TILE];
    __shared__ int      sDst[TILE];

    const int       lane = threadIdx.x;     // wave32
    const long long e0   = (long long)blockIdx.x * TILE;
    const int       c    = lane * 4;

    // prefetch a future chunk of the edge-index streams (global_prefetch_b8)
    __builtin_prefetch(src + e0 + 4096, 0, 0);
    __builtin_prefetch(dst + e0 + 4096, 0, 0);

#if USE_ASYNC_LDS
    // ---- stage: ASYNCcnt-tracked DMA, global -> LDS, no VGPR round-trip
#pragma unroll 4
    for (int r = 0; r < TILE; ++r) {
        long long e = e0 + r;
        int is = (e < E) ? src[e] : 0;
        int id = (e < E) ? dst[e] : 0;
        if (lane == r) sDst[r] = id;
        async_copy_b128(h + (long long)is * D + c, &sA[r][c]);
        async_copy_b128(h + (long long)id * D + c, &sB[r][c]);
    }
    wait_asynccnt0();
    __syncthreads();
    // convert LDS f32 rows to f16 rows for the WMMA operands
#pragma unroll 4
    for (int r = 0; r < TILE; ++r) {
        float4 va = *(const float4*)&sA[r][c];
        float4 vb = *(const float4*)&sB[r][c];
        sAh[r][c + 0] = (_Float16)va.x; sAh[r][c + 1] = (_Float16)va.y;
        sAh[r][c + 2] = (_Float16)va.z; sAh[r][c + 3] = (_Float16)va.w;
        sBh[r][c + 0] = (_Float16)vb.x; sBh[r][c + 1] = (_Float16)vb.y;
        sBh[r][c + 2] = (_Float16)vb.z; sBh[r][c + 3] = (_Float16)vb.w;
    }
#else
    // ---- fallback: coalesced 512B row reads through VGPRs
#pragma unroll 4
    for (int r = 0; r < TILE; ++r) {
        long long e = e0 + r;
        int is = (e < E) ? src[e] : 0;
        int id = (e < E) ? dst[e] : 0;
        if (lane == r) sDst[r] = id;
        float4 va = *(const float4*)(h + (long long)is * D + c);
        float4 vb = *(const float4*)(h + (long long)id * D + c);
        sA[r][c + 0] = va.x; sA[r][c + 1] = va.y; sA[r][c + 2] = va.z; sA[r][c + 3] = va.w;
        sAh[r][c + 0] = (_Float16)va.x; sAh[r][c + 1] = (_Float16)va.y;
        sAh[r][c + 2] = (_Float16)va.z; sAh[r][c + 3] = (_Float16)va.w;
        sBh[r][c + 0] = (_Float16)vb.x; sBh[r][c + 1] = (_Float16)vb.y;
        sBh[r][c + 2] = (_Float16)vb.z; sBh[r][c + 3] = (_Float16)vb.w;
    }
#endif
    __syncthreads();

    // ---- WMMA: acc(16x16) += A(16x32) * B(32x16), 4 steps over K=128
    const int half = lane >> 4;   // which 16-lane half
    const int idx  = lane & 15;   // M for A rows, N for B cols
    v8f acc = {};
#pragma unroll
    for (int kb = 0; kb < D; kb += 32) {
        V16 a, b;
        // A layout (16-bit A 16x32): lanes 0-15 hold K 0..7 / 16..23, lanes 16-31 K 8..15 / 24..31
        a.h[0] = *(const v8h*)(&sAh[idx][kb + 8 * half]);
        a.h[1] = *(const v8h*)(&sAh[idx][kb + 16 + 8 * half]);
        // B layout (32x16): same K order as A (lanes 0-15 K 0..7 / 16..23, lanes 16-31 K 8..15 / 24..31), N = lane%16
        b.h[0] = *(const v8h*)(&sBh[idx][kb + 8 * half]);
        b.h[1] = *(const v8h*)(&sBh[idx][kb + 16 + 8 * half]);
        acc = __builtin_amdgcn_wmma_f32_16x16x32_f16(
            /*neg_a=*/false, a.v, /*neg_b=*/false, b.v,
            /*c_mod=*/(short)0, acc, /*reuse_a=*/false, /*reuse_b=*/false);
    }

    // ---- extract diagonal (C layout: vgpr v, lanes 0-15: M=v,N=lane; lanes 16-31: M=v+8,N=lane-16)
#pragma unroll
    for (int v = 0; v < 8; ++v) sC[v][lane] = acc[v];
    __syncthreads();
    if (lane < TILE) {
        float d = (lane < 8) ? sC[lane][lane] : sC[lane - 8][lane + 16];
        sAlpha[lane] = fast_rcp(1.0f + __expf(-d));   // sigmoid via v_rcp_f32
    }
    __syncthreads();

    // ---- scatter: aggr[dst[e]] += alpha * h[src[e]]  (hardware f32 fadd atomics)
    for (int r = 0; r < TILE; ++r) {
        long long e = e0 + r;
        if (e >= E) break;
        float a = sAlpha[r];
        float* drow = aggr + (long long)sDst[r] * D;
        atomic_add_f32(drow + c + 0, a * sA[r][c + 0]);
        atomic_add_f32(drow + c + 1, a * sA[r][c + 1]);
        atomic_add_f32(drow + c + 2, a * sA[r][c + 2]);
        atomic_add_f32(drow + c + 3, a * sA[r][c + 3]);
    }
}

// ---------------- finalize hop: add sigma*noise, L2-normalize in place ----------------
__global__ __launch_bounds__(256) void finalize_kernel(float* __restrict__ buf, int N, unsigned seed) {
    int row  = blockIdx.x * 8 + (threadIdx.x >> 5);
    int lane = threadIdx.x & 31;
    if (row >= N) return;
    long long base = (long long)row * D + lane * 4;
    float4 v = *(float4*)(buf + base);
    unsigned u = (unsigned)base;
    v.x += SIGMA * gaussf(u + 0u, seed);
    v.y += SIGMA * gaussf(u + 1u, seed);
    v.z += SIGMA * gaussf(u + 2u, seed);
    v.w += SIGMA * gaussf(u + 3u, seed);
    float s = v.x * v.x + v.y * v.y + v.z * v.z + v.w * v.w;
#pragma unroll
    for (int off = 16; off > 0; off >>= 1) s += __shfl_xor(s, off);
    float inv = fast_rcp(fmaxf(sqrtf(s), EPS));
    v.x *= inv; v.y *= inv; v.z *= inv; v.w *= inv;
    *(float4*)(buf + base) = v;
}

extern "C" void kernel_launch(void* const* d_in, const int* in_sizes, int n_in,
                              void* d_out, int out_size, void* d_ws, size_t ws_size,
                              hipStream_t stream) {
    const float* x  = (const float*)d_in[0];
    const int*   ei = (const int*)d_in[1];
    const int N = in_sizes[0] / D;       // 50000
    const int E = in_sizes[1] / 2;       // 800000
    const int* src = ei;
    const int* dst = ei + E;
    float* out = (float*)d_out;          // [4, N, D]
    const long long ND = (long long)N * D;

    dim3 rowGrid((N + 7) / 8);           // 8 waves per 256-thread block, one row each

    // hop 0 output: normalized x
    norm_kernel<<<rowGrid, 256, 0, stream>>>(x, out, N);

    for (int k = 0; k < 3; ++k) {
        float* hk = out + (long long)k * ND;        // current h (read)
        float* hn = out + (long long)(k + 1) * ND;  // accumulate + finalize in place
        long long n4 = ND / 4;
        zero_kernel<<<(unsigned)((n4 + 255) / 256), 256, 0, stream>>>((float4*)hn, n4);
        edge_kernel<<<(E + TILE - 1) / TILE, 32, 0, stream>>>(hk, src, dst, hn, E);
        finalize_kernel<<<rowGrid, 256, 0, stream>>>(hn, N, (unsigned)(k + 1));
    }
}
